// ImprovedDecoder_42743514530554
// MI455X (gfx1250) — matrix operand order repaired and checked by reading, not compiled, not run on hardware
//
#include <hip/hip_runtime.h>
#include <hip/hip_bf16.h>

// ---------------------------------------------------------------------------
// ImprovedDecoder on MI455X (gfx1250, wave32, WMMA).
//
// Key structural facts exploited (see analysis):
//  * attention sequence length == 1  =>  alignments == 1.0, ctx == ce (const)
//  * prenet depends only on x_t      =>  precomputed for all T in parallel
//  * sequential core = 4 GEMVs/step  =>  WMMA f16 16x16x32 with replicated-A
//  * all weights (~27MB f16) live in the 192MB L2 across the 1000-step scan
// ---------------------------------------------------------------------------

typedef _Float16 v8h  __attribute__((ext_vector_type(8)));
typedef _Float16 v16h __attribute__((ext_vector_type(16)));
typedef float    v8f  __attribute__((ext_vector_type(8)));

#define NWG 32
#define TPB 256
#define WAVES_PER_BLK (TPB / 32)

__device__ inline v16h make16(v8h lo, v8h hi) {
  v16h r;
#pragma unroll
  for (int i = 0; i < 8; ++i) { r[i] = lo[i]; r[i + 8] = hi[i]; }
  return r;
}

// A fragment (16x32 f16, all 16 rows replicated with the same x vector).
// Layout (matches packed B): lanes<16 hold K {0..15}; lanes>=16 hold K {16..31}.
__device__ inline v16h load_a_tile(const _Float16* __restrict__ x, int kt, int lane) {
  const _Float16* p = x + kt * 32 + ((lane & 16) ? 16 : 0);
  return make16(*(const v8h*)p, *(const v8h*)(p + 8));
}

// B fragment (32x16 f16) from the pre-packed tile: lane holds 16 contiguous halves.
__device__ inline v16h load_b_tile(const _Float16* __restrict__ pw, int lane) {
  const _Float16* p = pw + lane * 16;
  return make16(*(const v8h*)p, *(const v8h*)(p + 8));
}

__device__ inline v8f wmma_f16(v16h a, v16h b, v8f c) {
  return __builtin_amdgcn_wmma_f32_16x16x32_f16(false, a, false, b, (short)0, c,
                                                false, false);
}

__device__ inline float sigm(float x) { return 1.0f / (1.0f + __expf(-x)); }

// Sense-counting grid barrier (count reset before generation bump).
__device__ inline void gbar(volatile int* cnt, volatile int* gen, int nblk) {
  __syncthreads();
  if (threadIdx.x == 0) {
    __threadfence();
    int g = *gen;
    if (atomicAdd((int*)cnt, 1) == nblk - 1) {
      *cnt = 0;
      __threadfence();
      atomicAdd((int*)gen, 1);
    } else {
      while (*gen == g) { __builtin_amdgcn_s_sleep(8); }
    }
    __threadfence();
  }
  __syncthreads();
}

// ---------------------------------------------------------------------------
// Pack a (4096 x K) f32 weight matrix (rows = outputs) into WMMA-B f16 tiles.
// Tile (nt, kt) = 512 halves; half index = lane*16 + i;
//   n = nt*16 + (lane&15);  k = kt*32 + i + ((lane&16) ? 16 : 0)
// ---------------------------------------------------------------------------
__global__ void pack_w16(const float* __restrict__ src, int ld, int colOff,
                         int KT, size_t total, _Float16* __restrict__ dst) {
  for (size_t idx = (size_t)blockIdx.x * blockDim.x + threadIdx.x; idx < total;
       idx += (size_t)gridDim.x * blockDim.x) {
    int b = (int)(idx & 511);
    size_t tk = idx >> 9;
    int kt = (int)(tk % KT);
    int nt = (int)(tk / KT);
    int lane = b >> 4, i = b & 15;
    int n = nt * 16 + (lane & 15);
    int k = kt * 32 + i + ((lane & 16) ? 16 : 0);
    dst[idx] = (_Float16)src[(size_t)n * ld + colOff + k];
  }
}

// ---------------------------------------------------------------------------
// Constant folds + state init + barrier reset (must run every launch: the
// harness does not re-zero the workspace between replays).
// ---------------------------------------------------------------------------
__global__ void prep_consts(const float* __restrict__ ce,
                            const float* __restrict__ Wih0,
                            const float* __restrict__ bih0, const float* __restrict__ bhh0,
                            const float* __restrict__ bih1, const float* __restrict__ bhh1,
                            const float* __restrict__ Wf, const float* __restrict__ bf,
                            const float* __restrict__ init_h, const float* __restrict__ init_c,
                            float* base0_t0, float* base0_rest, float* base1,
                            float* mel_const, float* c0f, float* c1f, float* h1f,
                            _Float16* h0h, _Float16* h1h, int* bar_cnt, int* bar_gen) {
  for (int i = blockIdx.x * blockDim.x + threadIdx.x; i < 4096;
       i += gridDim.x * blockDim.x) {
    float s = bih0[i] + bhh0[i];
    base0_t0[i] = s;
    float dot = 0.f;
    const float* wr = Wih0 + (size_t)i * 1280 + 256;
    for (int k = 0; k < 1024; ++k) dot += ce[k] * wr[k];
    base0_rest[i] = s + dot;
    base1[i] = bih1[i] + bhh1[i];
    if (i < 80) {
      float m = bf[i];
      const float* wf = Wf + (size_t)i * 2048 + 1024;
      for (int k = 0; k < 1024; ++k) m += ce[k] * wf[k];
      mel_const[i] = m;
    }
    if (i < 1024) {
      float h0 = init_h[i], h1 = init_h[1024 + i];
      c0f[i] = init_c[i];
      c1f[i] = init_c[1024 + i];
      h1f[i] = h1;
      h0h[i] = (_Float16)h0;
      h1h[i] = (_Float16)h1;
    }
    if (i == 0) { *bar_cnt = 0; *bar_gen = 0; }
  }
}

// ---------------------------------------------------------------------------
// Prenet for all timesteps (depends only on target_mels). One block per t.
// ---------------------------------------------------------------------------
__global__ __launch_bounds__(256) void prenet_all(
    const float* __restrict__ mels,                 // (T, 80)
    const float* __restrict__ pW0, const float* __restrict__ pb0,
    const float* __restrict__ pW1, const float* __restrict__ pb1,
    _Float16* __restrict__ preH, int T) {
  __shared__ float p0[256];
  int t = blockIdx.x;
  if (t >= T) return;
  int o = threadIdx.x;
  const float* x = mels + (size_t)t * 80;
  float a = pb0[o];
  const float* w0 = pW0 + (size_t)o * 80;
  for (int k = 0; k < 80; ++k) a += w0[k] * x[k];
  p0[o] = fmaxf(a, 0.f);
  __syncthreads();
  float b = pb1[o];
  const float* w1 = pW1 + (size_t)o * 256;
  for (int k = 0; k < 256; ++k) b += w1[k] * p0[k];
  preH[(size_t)t * 256 + o] = (_Float16)fmaxf(b, 0.f);
}

// ---------------------------------------------------------------------------
// Persistent sequential decoder: 32 blocks x 8 wave32 = 256 waves; one WMMA
// output tile (16 gates) per wave per GEMV phase; 4 grid barriers per step.
// ---------------------------------------------------------------------------
__global__ __launch_bounds__(TPB) void decoder_scan(
    const _Float16* __restrict__ Pih0, const _Float16* __restrict__ Phh0,
    const _Float16* __restrict__ Pih1, const _Float16* __restrict__ Phh1,
    const _Float16* __restrict__ preH,
    const float* __restrict__ base0_t0, const float* __restrict__ base0_rest,
    const float* __restrict__ base1, const float* __restrict__ mel_const,
    const float* __restrict__ Wf,
    float* c0f, float* c1f, float* h1f,
    _Float16* h0h, _Float16* h1h,
    float* gates0, float* gates1, float* mel_out, int T,
    int* bar_cnt, int* bar_gen) {
  const int lane = threadIdx.x & 31;
  const int wid  = threadIdx.x >> 5;
  const int tile = blockIdx.x * WAVES_PER_BLK + wid;   // 0..255
  const int gtid = blockIdx.x * TPB + threadIdx.x;

  for (int t = 0; t < T; ++t) {
    // -------- Phase A: gates0 = pre1 @ Wih0[:, :256].T + h0 @ Whh0.T + base0
    {
      v8f acc = {};
      const _Float16* xa = preH + (size_t)t * 256;
      const _Float16* pb = Pih0 + (size_t)tile * 8 * 512;
#pragma unroll 2
      for (int kt = 0; kt < 8; ++kt) {
        __builtin_prefetch(pb + 512 + lane * 16, 0, 1);
        v16h a = load_a_tile(xa, kt, lane);
        v16h b = load_b_tile(pb, lane);
        acc = wmma_f16(a, b, acc);
        pb += 512;
      }
      const _Float16* ph = Phh0 + (size_t)tile * 32 * 512;
#pragma unroll 2
      for (int kt = 0; kt < 32; ++kt) {
        __builtin_prefetch(ph + 512 + lane * 16, 0, 1);
        v16h a = load_a_tile(h0h, kt, lane);
        v16h b = load_b_tile(ph, lane);
        acc = wmma_f16(a, b, acc);
        ph += 512;
      }
      const float* base = (t == 0) ? base0_t0 : base0_rest;
      if (lane < 16) {
        int n = tile * 16 + lane;
        gates0[n] = acc[0] + base[n];
      }
      // mel[t-1] = h1 @ Wf[:, :1024].T + mel_const   (h1 is step-(t-1) state)
      if (blockIdx.x == NWG - 1 && threadIdx.x < 80 && t > 0) {
        float s = mel_const[threadIdx.x];
        const float* wr = Wf + (size_t)threadIdx.x * 2048;
        for (int k = 0; k < 1024; ++k) s += h1f[k] * wr[k];
        mel_out[(size_t)(t - 1) * 80 + threadIdx.x] = s;
      }
    }
    gbar(bar_cnt, bar_gen, NWG);

    // -------- Phase B: LSTM0 elementwise
    if (gtid < 1024) {
      float ii = gates0[gtid], ff = gates0[1024 + gtid];
      float gg = gates0[2048 + gtid], oo = gates0[3072 + gtid];
      float c = sigm(ff) * c0f[gtid] + sigm(ii) * tanhf(gg);
      float h = sigm(oo) * tanhf(c);
      c0f[gtid] = c;
      h0h[gtid] = (_Float16)h;
    }
    gbar(bar_cnt, bar_gen, NWG);

    // -------- Phase C: gates1 = h0 @ Wih1.T + h1 @ Whh1.T + base1
    {
      v8f acc = {};
      const _Float16* pi = Pih1 + (size_t)tile * 32 * 512;
#pragma unroll 2
      for (int kt = 0; kt < 32; ++kt) {
        __builtin_prefetch(pi + 512 + lane * 16, 0, 1);
        v16h a = load_a_tile(h0h, kt, lane);
        v16h b = load_b_tile(pi, lane);
        acc = wmma_f16(a, b, acc);
        pi += 512;
      }
      const _Float16* ph = Phh1 + (size_t)tile * 32 * 512;
#pragma unroll 2
      for (int kt = 0; kt < 32; ++kt) {
        __builtin_prefetch(ph + 512 + lane * 16, 0, 1);
        v16h a = load_a_tile(h1h, kt, lane);
        v16h b = load_b_tile(ph, lane);
        acc = wmma_f16(a, b, acc);
        ph += 512;
      }
      if (lane < 16) {
        int n = tile * 16 + lane;
        gates1[n] = acc[0] + base1[n];
      }
    }
    gbar(bar_cnt, bar_gen, NWG);

    // -------- Phase D: LSTM1 elementwise
    if (gtid < 1024) {
      float ii = gates1[gtid], ff = gates1[1024 + gtid];
      float gg = gates1[2048 + gtid], oo = gates1[3072 + gtid];
      float c = sigm(ff) * c1f[gtid] + sigm(ii) * tanhf(gg);
      float h = sigm(oo) * tanhf(c);
      c1f[gtid] = c;
      h1f[gtid] = h;
      h1h[gtid] = (_Float16)h;
    }
    gbar(bar_cnt, bar_gen, NWG);
  }

  // Final mel[T-1]
  if (blockIdx.x == NWG - 1 && threadIdx.x < 80) {
    float s = mel_const[threadIdx.x];
    const float* wr = Wf + (size_t)threadIdx.x * 2048;
    for (int k = 0; k < 1024; ++k) s += h1f[k] * wr[k];
    mel_out[(size_t)(T - 1) * 80 + threadIdx.x] = s;
  }
}

// ---------------------------------------------------------------------------
// Postnet conv (kernel 5, pad 2) + batchnorm (+ optional tanh).
// Output is channel-major [Cout][T]; input layout via strides.
// ---------------------------------------------------------------------------
__global__ void postconv(const float* __restrict__ in, int inSC, int inST,
                         const float* __restrict__ W, const float* __restrict__ b,
                         const float* __restrict__ g, const float* __restrict__ bb,
                         const float* __restrict__ rm, const float* __restrict__ rv,
                         int Cin, int Cout, int T, float* __restrict__ out,
                         int doTanh) {
  int idx = blockIdx.x * blockDim.x + threadIdx.x;
  int total = Cout * T;
  if (idx >= total) return;
  int co = idx / T, t = idx % T;
  float acc = b[co];
  const float* wbase = W + (size_t)co * Cin * 5;
  for (int ci = 0; ci < Cin; ++ci) {
    const float* wr = wbase + ci * 5;
    const float* xr = in + (size_t)ci * inSC;
#pragma unroll
    for (int k = 0; k < 5; ++k) {
      int tt = t + k - 2;
      if (tt >= 0 && tt < T) acc += wr[k] * xr[(size_t)tt * inST];
    }
  }
  float inv = rsqrtf(rv[co] + 1e-5f);
  float y = (acc - rm[co]) * (g[co] * inv) + bb[co];
  if (doTanh) y = tanhf(y);
  out[(size_t)co * T + t] = y;
}

// out = mel + postnet (transposed back), alignments = 1.0
__global__ void finalize(const float* __restrict__ mel,
                         const float* __restrict__ post, float* __restrict__ out,
                         int T, int M) {
  int idx = blockIdx.x * blockDim.x + threadIdx.x;
  int tm = T * M;
  if (idx < tm) {
    int t = idx / M, m = idx % M;
    out[idx] = mel[idx] + post[(size_t)m * T + t];
  } else if (idx < tm + T) {
    out[idx] = 1.0f;
  }
}

// ---------------------------------------------------------------------------
extern "C" void kernel_launch(void* const* d_in, const int* in_sizes, int n_in,
                              void* d_out, int out_size, void* d_ws, size_t ws_size,
                              hipStream_t stream) {
  (void)in_sizes; (void)n_in; (void)out_size; (void)ws_size;
  const int T = 1000, M = 80;

  const float* ce     = (const float*)d_in[0];
  const float* mels   = (const float*)d_in[1];
  const float* pW0    = (const float*)d_in[2];
  const float* pb0    = (const float*)d_in[3];
  const float* pW1    = (const float*)d_in[4];
  const float* pb1    = (const float*)d_in[5];
  const float* Wih0   = (const float*)d_in[11];
  const float* Whh0   = (const float*)d_in[12];
  const float* bih0   = (const float*)d_in[13];
  const float* bhh0   = (const float*)d_in[14];
  const float* Wih1   = (const float*)d_in[15];
  const float* Whh1   = (const float*)d_in[16];
  const float* bih1   = (const float*)d_in[17];
  const float* bhh1   = (const float*)d_in[18];
  const float* Wf     = (const float*)d_in[19];
  const float* bf     = (const float*)d_in[20];
  const float* init_h = (const float*)d_in[21];
  const float* init_c = (const float*)d_in[22];

  // Workspace carve-up (256B aligned)
  char* w = (char*)d_ws;
  auto carve = [&](size_t bytes) {
    char* p = w;
    w += (bytes + 255) & ~(size_t)255;
    return p;
  };
  _Float16* Pih0 = (_Float16*)carve((size_t)4096 * 256 * 2);
  _Float16* Phh0 = (_Float16*)carve((size_t)4096 * 1024 * 2);
  _Float16* Pih1 = (_Float16*)carve((size_t)4096 * 1024 * 2);
  _Float16* Phh1 = (_Float16*)carve((size_t)4096 * 1024 * 2);
  _Float16* preH = (_Float16*)carve((size_t)T * 256 * 2);
  float* base0_t0   = (float*)carve(4096 * 4);
  float* base0_rest = (float*)carve(4096 * 4);
  float* base1      = (float*)carve(4096 * 4);
  float* mel_const  = (float*)carve(80 * 4);
  float* c0f = (float*)carve(1024 * 4);
  float* c1f = (float*)carve(1024 * 4);
  float* h1f = (float*)carve(1024 * 4);
  _Float16* h0h = (_Float16*)carve(1024 * 2);
  _Float16* h1h = (_Float16*)carve(1024 * 2);
  float* gates0  = (float*)carve(4096 * 4);
  float* gates1  = (float*)carve(4096 * 4);
  float* mel_out = (float*)carve((size_t)T * M * 4);
  float* postA   = (float*)carve((size_t)512 * T * 4);
  float* postB   = (float*)carve((size_t)512 * T * 4);
  int* bar_cnt = (int*)carve(4);
  int* bar_gen = (int*)carve(4);

  // 1) Repack recurrent weights into WMMA-B f16 tiles (one-time-ish; cheap).
  pack_w16<<<2048, 256, 0, stream>>>(Wih0, 1280, 0, 8,  (size_t)4096 * 256,  Pih0);
  pack_w16<<<2048, 256, 0, stream>>>(Whh0, 1024, 0, 32, (size_t)4096 * 1024, Phh0);
  pack_w16<<<2048, 256, 0, stream>>>(Wih1, 1024, 0, 32, (size_t)4096 * 1024, Pih1);
  pack_w16<<<2048, 256, 0, stream>>>(Whh1, 1024, 0, 32, (size_t)4096 * 1024, Phh1);

  // 2) Constant folds (attention is constant), state init, barrier reset.
  prep_consts<<<16, 256, 0, stream>>>(ce, Wih0, bih0, bhh0, bih1, bhh1, Wf, bf,
                                      init_h, init_c, base0_t0, base0_rest,
                                      base1, mel_const, c0f, c1f, h1f, h0h, h1h,
                                      bar_cnt, bar_gen);

  // 3) Prenet for all timesteps in parallel.
  prenet_all<<<T, 256, 0, stream>>>(mels, pW0, pb0, pW1, pb1, preH, T);

  // 4) Persistent WMMA sequential decoder.
  decoder_scan<<<NWG, TPB, 0, stream>>>(Pih0, Phh0, Pih1, Phh1, preH, base0_t0,
                                        base0_rest, base1, mel_const, Wf, c0f,
                                        c1f, h1f, h0h, h1h, gates0, gates1,
                                        mel_out, T, bar_cnt, bar_gen);

  // 5) Postnet: 5 conv+BN layers.  chans = [80,512,512,512,512,80]
  const float* pw[5]; const float* pbv[5]; const float* gg[5];
  const float* bb[5]; const float* rm[5]; const float* rv[5];
  for (int i = 0; i < 5; ++i) {
    int base = 23 + 6 * i;
    pw[i]  = (const float*)d_in[base + 0];
    pbv[i] = (const float*)d_in[base + 1];
    gg[i]  = (const float*)d_in[base + 2];
    bb[i]  = (const float*)d_in[base + 3];
    rm[i]  = (const float*)d_in[base + 4];
    rv[i]  = (const float*)d_in[base + 5];
  }
  int blk512 = (512 * T + 255) / 256;
  int blk80  = (80 * T + 255) / 256;
  // layer 0: input is mel_out in (T, M) layout -> strides (C:1, T:M)
  postconv<<<blk512, 256, 0, stream>>>(mel_out, 1, M, pw[0], pbv[0], gg[0], bb[0],
                                       rm[0], rv[0], 80, 512, T, postA, 1);
  postconv<<<blk512, 256, 0, stream>>>(postA, T, 1, pw[1], pbv[1], gg[1], bb[1],
                                       rm[1], rv[1], 512, 512, T, postB, 1);
  postconv<<<blk512, 256, 0, stream>>>(postB, T, 1, pw[2], pbv[2], gg[2], bb[2],
                                       rm[2], rv[2], 512, 512, T, postA, 1);
  postconv<<<blk512, 256, 0, stream>>>(postA, T, 1, pw[3], pbv[3], gg[3], bb[3],
                                       rm[3], rv[3], 512, 512, T, postB, 1);
  postconv<<<blk80, 256, 0, stream>>>(postB, T, 1, pw[4], pbv[4], gg[4], bb[4],
                                      rm[4], rv[4], 512, 80, T, postA, 0);

  // 6) out = mel + postnet; alignments = ones.
  int blkF = (T * M + T + 255) / 256;
  finalize<<<blkF, 256, 0, stream>>>(mel_out, postA, (float*)d_out, T, M);
}
